// model_44143673868302
// MI455X (gfx1250) — matrix operand order repaired and checked by reading, not compiled, not run on hardware
//
#include <hip/hip_runtime.h>
#include <hip/hip_bf16.h>
#include <math.h>

typedef __bf16 bf16;
typedef __attribute__((ext_vector_type(16))) __bf16 v16bf;
typedef __attribute__((ext_vector_type(8)))  __bf16 v8bf;
typedef __attribute__((ext_vector_type(8)))  float  v8f;

#define TT   400
#define JJ   48
#define BB   64
#define DD   256
#define II   300
#define IPAD 320
#define NG   1024   // 4*D (gate dim)
#define H2   512    // 2*D
#define G8   2048   // 8*D

__device__ __forceinline__ float bf2f(bf16 x) { return (float)x; }
__device__ __forceinline__ bf16  f2bf(float x) { return (bf16)x; }

#if __has_builtin(__builtin_amdgcn_global_load_async_to_lds_b128)
#define ASYNC_LDS 1
#else
#define ASYNC_LDS 0
#endif

// Copy 16 bf16 (32B) global -> LDS. Async (CDNA5 GLOBAL_LOAD_ASYNC_TO_LDS_B128,
// tracked by ASYNCcnt) when available, else sync fallback through VGPRs.
__device__ __forceinline__ void stage16(const bf16* __restrict__ g, bf16* l) {
#if ASYNC_LDS
    typedef int v4i __attribute__((vector_size(16)));
    typedef __attribute__((address_space(1))) v4i* gp_t;
    typedef __attribute__((address_space(3))) v4i* lp_t;
    __builtin_amdgcn_global_load_async_to_lds_b128((gp_t)g,       (lp_t)l,       0, 0);
    __builtin_amdgcn_global_load_async_to_lds_b128((gp_t)(g + 8), (lp_t)(l + 8), 0, 0);
#else
    *(v8bf*)l       = *(const v8bf*)g;
    *(v8bf*)(l + 8) = *(const v8bf*)(g + 8);
#endif
}

__device__ __forceinline__ void async_wait0() {
#if ASYNC_LDS
#if __has_builtin(__builtin_amdgcn_s_wait_asynccnt)
    __builtin_amdgcn_s_wait_asynccnt(0);
#else
    asm volatile("s_wait_asynccnt 0x0" ::: "memory");
#endif
#endif
}

// ---------------------------------------------------------------- ctrl init
__global__ void k_init(unsigned* ctrl) {
    if (threadIdx.x < 64) ctrl[threadIdx.x] = 0u;
}

// ------------------------------------------------- fp32 -> bf16 (K-padded)
__global__ void k_cvtpad(const float* __restrict__ src, bf16* __restrict__ dst,
                         int rows, int cols, int colsPad) {
    size_t total = (size_t)rows * colsPad;
    for (size_t i = (size_t)blockIdx.x * blockDim.x + threadIdx.x; i < total;
         i += (size_t)gridDim.x * blockDim.x) {
        size_t r = i / colsPad;
        int    c = (int)(i % colsPad);
        dst[i] = (c < cols) ? f2bf(src[r * cols + c]) : f2bf(0.0f);
    }
}

// --------------------------------------------------------- device barrier
__device__ __forceinline__ void grid_barrier(unsigned* cnt, unsigned* gen, unsigned n) {
    __syncthreads();
    if (threadIdx.x == 0) {
        __threadfence();
        unsigned g = __hip_atomic_load(gen, __ATOMIC_RELAXED, __HIP_MEMORY_SCOPE_AGENT);
        if (__hip_atomic_fetch_add(cnt, 1u, __ATOMIC_ACQ_REL, __HIP_MEMORY_SCOPE_AGENT) == n - 1u) {
            __hip_atomic_store(cnt, 0u, __ATOMIC_RELAXED, __HIP_MEMORY_SCOPE_AGENT);
            __hip_atomic_fetch_add(gen, 1u, __ATOMIC_RELEASE, __HIP_MEMORY_SCOPE_AGENT);
        } else {
            while (__hip_atomic_load(gen, __ATOMIC_ACQUIRE, __HIP_MEMORY_SCOPE_AGENT) == g)
                __builtin_amdgcn_s_sleep(2);
        }
        __threadfence();
    }
    __syncthreads();
}

// --------------------------------------------------------------- WMMA GEMM
// C[M,N](bf16) = A[M,Kp](bf16) @ W[N,Kp](bf16)^T + bias0 + bias1
// Block: 256 thr (8 waves), tile 64(M) x 256(N). M%64==0, N%256==0, Kp%64==0.
// A is staged through a double-buffered LDS tile (64x64 per chunk) with
// async global->LDS copies; B streams from L2.
__global__ __launch_bounds__(256) void k_gemm(const bf16* __restrict__ A,
                                              const bf16* __restrict__ W,
                                              bf16* __restrict__ C,
                                              const float* __restrict__ bias0,
                                              const float* __restrict__ bias1,
                                              int M, int N, int Kp) {
    __shared__ bf16 sA[2][64 * 64];
    const int nblks = N >> 8;
    const int mblk  = blockIdx.x / nblks;
    const int nblk  = blockIdx.x % nblks;
    const int tid   = threadIdx.x;
    const int wave  = tid >> 5, lane = tid & 31;
    const int half  = lane >> 4, l15 = lane & 15;
    const int mbase = mblk * 64, nbase = nblk * 256;
    const int sr = tid >> 2, sc = (tid & 3) * 16;   // staging coords (16 elems/thread)

    v8f acc[4][2];
#pragma unroll
    for (int mt = 0; mt < 4; ++mt)
#pragma unroll
        for (int nt = 0; nt < 2; ++nt)
#pragma unroll
            for (int e = 0; e < 8; ++e) acc[mt][nt][e] = 0.0f;

    const bf16* Wl  = W + (size_t)nbase * Kp;
    const bf16* Ast = A + (size_t)(mbase + sr) * Kp + sc;

    // prologue: stage K-chunk 0
    stage16(Ast, &sA[0][sr * 64 + sc]);
    async_wait0();
    __syncthreads();

    const int nchunks = Kp >> 6;
    for (int ch = 0; ch < nchunks; ++ch) {
        const int cur = ch & 1, nxt = cur ^ 1;
        const int kb  = ch << 6;
        if (ch + 1 < nchunks)
            stage16(Ast + (ch + 1) * 64, &sA[nxt][sr * 64 + sc]);

        if (kb + 64 < Kp)
            __builtin_prefetch(Wl + (size_t)(wave * 32 + l15) * Kp + kb + 64, 0, 1);

#pragma unroll
        for (int ks = 0; ks < 64; ks += 32) {
            v16bf bfr[2];
#pragma unroll
            for (int nt = 0; nt < 2; ++nt) {
                int row = wave * 32 + nt * 16 + l15;
                bfr[nt] = *(const v16bf*)(Wl + (size_t)row * Kp + kb + ks + half * 16);
            }
#pragma unroll
            for (int mt = 0; mt < 4; ++mt) {
                v8bf alo = *(const v8bf*)&sA[cur][(mt * 16 + l15) * 64 + ks + half * 16];
                v8bf ahi = *(const v8bf*)&sA[cur][(mt * 16 + l15) * 64 + ks + half * 16 + 8];
                v16bf afr;
#pragma unroll
                for (int i = 0; i < 8; ++i) { afr[i] = alo[i]; afr[i + 8] = ahi[i]; }
                acc[mt][0] = __builtin_amdgcn_wmma_f32_16x16x32_bf16(false, afr, false, bfr[0],
                                                                     (short)0, acc[mt][0], false, false);
                acc[mt][1] = __builtin_amdgcn_wmma_f32_16x16x32_bf16(false, afr, false, bfr[1],
                                                                     (short)0, acc[mt][1], false, false);
            }
        }
        async_wait0();
        __syncthreads();
    }

#pragma unroll
    for (int nt = 0; nt < 2; ++nt) {
        int ng = nbase + wave * 32 + nt * 16 + l15;
        float bs = (bias0 ? bias0[ng] : 0.0f) + (bias1 ? bias1[ng] : 0.0f);
#pragma unroll
        for (int mt = 0; mt < 4; ++mt)
#pragma unroll
            for (int r = 0; r < 8; ++r) {
                int m = mbase + mt * 16 + half * 8 + r;
                C[(size_t)m * N + ng] = f2bf(acc[mt][nt][r] + bs);
            }
    }
}

// --------------------------------------------------- persistent LSTM layer
// grid = 8 blocks: dir = blk/4, slice = blk%4 (64 d-cols, all 4 gates).
// xp  : [2][Tl*BB][NG] bf16 (already has b_ih+b_hh)
// Whh : [2][NG][DD] bf16, Hout: [Tl][BB][H2] bf16 (dir -> column offset DD)
// hstate: [2 dirs][2 bufs][BB][DD] bf16, cstate: [2][BB][DD] f32
__global__ __launch_bounds__(256) void k_lstm(const bf16* __restrict__ xp,
                                              const bf16* __restrict__ Whh,
                                              bf16* __restrict__ Hout,
                                              bf16* __restrict__ hstate,
                                              float* __restrict__ cstate,
                                              unsigned* __restrict__ ctrl,
                                              int Tl) {
    __shared__ float sg[BB * 256];   // gates for this slice: [b][g*64+j]
    const int dir = blockIdx.x >> 2;
    const int sl  = blockIdx.x & 3;
    const int tid = threadIdx.x;
    const int wave = tid >> 5, lane = tid & 31;
    const int half = lane >> 4, l15 = lane & 15;

    const bf16* Wd  = Whh + (size_t)dir * NG * DD;
    const bf16* xpd = xp + (size_t)dir * Tl * BB * NG;
    bf16*  hb = hstate + (size_t)dir * 2 * BB * DD;
    float* cd = cstate + (size_t)dir * BB * DD;
    unsigned* cnt = ctrl + dir * 2;
    unsigned* gen = ctrl + dir * 2 + 1;

    // zero this slice's state (both h buffers + c)
    for (int i = tid; i < BB * 64; i += 256) {
        int b = i >> 6, j = i & 63;
        hb[b * DD + sl * 64 + j]           = f2bf(0.0f);
        hb[BB * DD + b * DD + sl * 64 + j] = f2bf(0.0f);
        cd[b * DD + sl * 64 + j]           = 0.0f;
    }
    grid_barrier(cnt, gen, 4u);

    for (int t = 0; t < Tl; ++t) {
        const int te = dir ? (Tl - 1 - t) : t;
        const bf16* hprev = hb + (size_t)(t & 1) * BB * DD;
        bf16*       hnext = hb + (size_t)((t + 1) & 1) * BB * DD;

        v8f acc[4][2];
#pragma unroll
        for (int mt = 0; mt < 4; ++mt)
#pragma unroll
            for (int nt = 0; nt < 2; ++nt)
#pragma unroll
                for (int e = 0; e < 8; ++e) acc[mt][nt][e] = 0.0f;

        // gates(64 x 256_local) = h_prev(64x256) @ Whh_slice^T
        for (int kb = 0; kb < DD; kb += 32) {
            v16bf bfr[2];
#pragma unroll
            for (int nt = 0; nt < 2; ++nt) {
                int nl  = wave * 32 + nt * 16 + l15;
                int row = (nl >> 6) * DD + sl * 64 + (nl & 63); // gate*256 + slice*64 + j
                bfr[nt] = *(const v16bf*)(Wd + (size_t)row * DD + kb + half * 16);
            }
#pragma unroll
            for (int mt = 0; mt < 4; ++mt) {
                v8bf alo = *(const v8bf*)(hprev + (mt * 16 + l15) * DD + kb + half * 16);
                v8bf ahi = *(const v8bf*)(hprev + (mt * 16 + l15) * DD + kb + half * 16 + 8);
                v16bf afr;
#pragma unroll
                for (int i = 0; i < 8; ++i) { afr[i] = alo[i]; afr[i + 8] = ahi[i]; }
                acc[mt][0] = __builtin_amdgcn_wmma_f32_16x16x32_bf16(false, afr, false, bfr[0],
                                                                     (short)0, acc[mt][0], false, false);
                acc[mt][1] = __builtin_amdgcn_wmma_f32_16x16x32_bf16(false, afr, false, bfr[1],
                                                                     (short)0, acc[mt][1], false, false);
            }
        }

        // stash gates + xp into LDS
        const bf16* xpt = xpd + (size_t)te * BB * NG;
#pragma unroll
        for (int mt = 0; mt < 4; ++mt)
#pragma unroll
            for (int nt = 0; nt < 2; ++nt) {
                int nl  = wave * 32 + nt * 16 + l15;
                int col = (nl >> 6) * DD + sl * 64 + (nl & 63);
#pragma unroll
                for (int r = 0; r < 8; ++r) {
                    int m = mt * 16 + half * 8 + r;
                    sg[m * 256 + nl] = acc[mt][nt][r] + bf2f(xpt[(size_t)m * NG + col]);
                }
            }
        __syncthreads();

        // LSTM cell elementwise for this slice
        for (int i = tid; i < BB * 64; i += 256) {
            int b = i >> 6, j = i & 63;
            float gi = sg[b * 256 + j];
            float gf = sg[b * 256 + 64 + j];
            float gg = sg[b * 256 + 128 + j];
            float go = sg[b * 256 + 192 + j];
            float c  = cd[b * DD + sl * 64 + j];
            float si = 1.0f / (1.0f + __expf(-gi));
            float sf = 1.0f / (1.0f + __expf(-gf));
            float so = 1.0f / (1.0f + __expf(-go));
            float cn = sf * c + si * tanhf(gg);
            float hn = so * tanhf(cn);
            cd[b * DD + sl * 64 + j] = cn;
            hnext[b * DD + sl * 64 + j] = f2bf(hn);
            Hout[((size_t)te * BB + b) * H2 + dir * DD + sl * 64 + j] = f2bf(hn);
        }
        grid_barrier(cnt, gen, 4u);
    }
}

// ------------------------------------------------------------- U @ w2 dots
__global__ void k_uw2(const bf16* __restrict__ U, const float* __restrict__ attw,
                      float* __restrict__ Uw2) {
    const float* w2 = attw + H2;
    for (int i = blockIdx.x * blockDim.x + threadIdx.x; i < JJ * BB;
         i += blockDim.x * gridDim.x) {
        int j = i / BB, b = i % BB;
        float s = 0.0f;
        const bf16* Ur = U + ((size_t)j * BB + b) * H2;
        for (int d = 0; d < H2; ++d) s += bf2f(Ur[d]) * w2[d];
        Uw2[i] = s;
    }
}

// ------------------ fused S / row-max(bw) / softmax / U_tilda, per t-block
__global__ __launch_bounds__(256) void k_attn(const bf16* __restrict__ H,
                                              const bf16* __restrict__ U,
                                              const float* __restrict__ Uw2,
                                              const float* __restrict__ attw,
                                              const float* __restrict__ attb,
                                              float* __restrict__ bw,
                                              bf16* __restrict__ Util) {
    __shared__ float ss[BB][JJ];
    __shared__ float shw1[BB];
    __shared__ float sinv[BB];
    const int t = blockIdx.x, tid = threadIdx.x;
    const int b = tid >> 2, part = tid & 3;
    const bf16* Hrow = H + ((size_t)t * BB + b) * H2;
    const float* w1 = attw;
    const float* w3 = attw + 2 * H2;

    if (part == 0) {
        float s = 0.0f;
        for (int d = 0; d < H2; ++d) s += bf2f(Hrow[d]) * w1[d];
        shw1[b] = s;
    }
    __syncthreads();
    const float ab = attb[0];
    for (int j = part; j < JJ; j += 4) {
        const bf16* Urow = U + ((size_t)j * BB + b) * H2;
        float s = 0.0f;
        for (int d = 0; d < H2; ++d) s += bf2f(Hrow[d]) * w3[d] * bf2f(Urow[d]);
        ss[b][j] = shw1[b] + Uw2[j * BB + b] + s + ab;
    }
    __syncthreads();
    if (part == 0) {
        float mx = ss[b][0];
        for (int j = 1; j < JJ; ++j) mx = fmaxf(mx, ss[b][j]);
        bw[(size_t)t * BB + b] = mx;
        float sum = 0.0f;
        for (int j = 0; j < JJ; ++j) { float e = __expf(ss[b][j] - mx); ss[b][j] = e; sum += e; }
        sinv[b] = 1.0f / sum;
    }
    __syncthreads();
    const float inv = sinv[b];
    for (int d = part * 128; d < part * 128 + 128; ++d) {
        float s = 0.0f;
        for (int j = 0; j < JJ; ++j) s += ss[b][j] * bf2f(U[((size_t)j * BB + b) * H2 + d]);
        Util[((size_t)t * BB + b) * H2 + d] = f2bf(s * inv);
    }
}

// ------------------------------------------------------------- h_tilda[b,d]
__global__ void k_htil(const bf16* __restrict__ H, const float* __restrict__ bw,
                       bf16* __restrict__ htil) {
    const int b = blockIdx.x, d = threadIdx.x; // 512 threads
    float s = 0.0f;
    for (int t = 0; t < TT; ++t)
        s += bw[(size_t)t * BB + b] * bf2f(H[((size_t)t * BB + b) * H2 + d]);
    htil[b * H2 + d] = f2bf(s);
}

// ------------------------------------------------------ Gin = concat(...)
__global__ void k_gin(const bf16* __restrict__ H, const bf16* __restrict__ Util,
                      const bf16* __restrict__ htil, bf16* __restrict__ Gin) {
    size_t total = (size_t)TT * BB * H2;
    for (size_t i = (size_t)blockIdx.x * blockDim.x + threadIdx.x; i < total;
         i += (size_t)gridDim.x * blockDim.x) {
        size_t row = i / H2;
        int d = (int)(i % H2), b = (int)(row % BB);
        float h = bf2f(H[i]), u = bf2f(Util[i]), ht = bf2f(htil[b * H2 + d]);
        bf16* g = Gin + row * G8;
        g[d]          = f2bf(h);
        g[H2 + d]     = f2bf(u);
        g[2 * H2 + d] = f2bf(h * u);
        g[3 * H2 + d] = f2bf(h * ht);
    }
}

// ---------------------------------- p = [G|M] @ ow + ob, softmax over batch
__global__ __launch_bounds__(256) void k_pout(const bf16* __restrict__ G,
                                              const bf16* __restrict__ M,
                                              const float* __restrict__ ow,
                                              const float* __restrict__ ob,
                                              float* __restrict__ out) {
    __shared__ float sp[BB][4];
    __shared__ float sv[BB];
    __shared__ float sinv;
    const int t = blockIdx.x, tid = threadIdx.x;
    const int b = tid >> 2, part = tid & 3;
    const bf16* Grow = G + ((size_t)t * BB + b) * G8;
    const bf16* Mrow = M + ((size_t)t * BB + b) * H2;
    float s = 0.0f;
    for (int k = part * 640; k < part * 640 + 640; ++k) {
        float v = (k < G8) ? bf2f(Grow[k]) : bf2f(Mrow[k - G8]);
        s += v * ow[k];
    }
    sp[b][part] = s;
    __syncthreads();
    if (tid < BB) sv[tid] = sp[tid][0] + sp[tid][1] + sp[tid][2] + sp[tid][3] + ob[0];
    __syncthreads();
    if (tid == 0) {
        float mx = sv[0];
        for (int i = 1; i < BB; ++i) mx = fmaxf(mx, sv[i]);
        float sum = 0.0f;
        for (int i = 0; i < BB; ++i) { float e = __expf(sv[i] - mx); sv[i] = e; sum += e; }
        sinv = 1.0f / sum;
    }
    __syncthreads();
    if (tid < BB) out[(size_t)t * BB + tid] = sv[tid] * sinv;
}

// ============================================================== host side
static inline size_t aup(size_t x) { return (x + 255) & ~(size_t)255; }

extern "C" void kernel_launch(void* const* d_in, const int* in_sizes, int n_in,
                              void* d_out, int out_size, void* d_ws, size_t ws_size,
                              hipStream_t stream) {
    (void)in_sizes; (void)n_in; (void)out_size; (void)ws_size;
    const float* X      = (const float*)d_in[0];
    const float* Q      = (const float*)d_in[1];
    const float* cW_ih  = (const float*)d_in[2];
    const float* cW_hh  = (const float*)d_in[3];
    const float* cb_ih  = (const float*)d_in[4];
    const float* cb_hh  = (const float*)d_in[5];
    const float* qW_ih  = (const float*)d_in[6];
    const float* qW_hh  = (const float*)d_in[7];
    const float* qb_ih  = (const float*)d_in[8];
    const float* qb_hh  = (const float*)d_in[9];
    const float* m1W_ih = (const float*)d_in[10];
    const float* m1W_hh = (const float*)d_in[11];
    const float* m1b_ih = (const float*)d_in[12];
    const float* m1b_hh = (const float*)d_in[13];
    const float* m2W_ih = (const float*)d_in[14];
    const float* m2W_hh = (const float*)d_in[15];
    const float* m2b_ih = (const float*)d_in[16];
    const float* m2b_hh = (const float*)d_in[17];
    const float* att_w  = (const float*)d_in[18];
    const float* att_b  = (const float*)d_in[19];
    const float* att1_w = (const float*)d_in[20];
    const float* att1_b = (const float*)d_in[21];
    const float* o1_w   = (const float*)d_in[22];
    const float* o1_b   = (const float*)d_in[23];
    const float* o2_w   = (const float*)d_in[24];
    const float* o2_b   = (const float*)d_in[25];
    float* out = (float*)d_out;

    char* ws = (char*)d_ws;
    size_t off = 0;
    auto take = [&](size_t bytes) { char* p = ws + off; off = aup(off + bytes); return p; };

    unsigned* ctrl = (unsigned*)take(64 * sizeof(unsigned));
    bf16* Xpad = (bf16*)take((size_t)TT * BB * IPAD * 2);
    bf16* Qpad = (bf16*)take((size_t)JJ * BB * IPAD * 2);
    bf16* WihC = (bf16*)take((size_t)2 * NG * IPAD * 2);
    bf16* WihQ = (bf16*)take((size_t)2 * NG * IPAD * 2);
    bf16* WhhC = (bf16*)take((size_t)2 * NG * DD * 2);
    bf16* WhhQ = (bf16*)take((size_t)2 * NG * DD * 2);
    bf16* Whh1 = (bf16*)take((size_t)2 * NG * DD * 2);
    bf16* Whh2 = (bf16*)take((size_t)2 * NG * DD * 2);
    bf16* Wih1 = (bf16*)take((size_t)2 * NG * G8 * 2);
    bf16* Wih2 = (bf16*)take((size_t)2 * NG * H2 * 2);
    bf16* A1W  = (bf16*)take((size_t)G8 * G8 * 2);
    bf16* xpB  = (bf16*)take((size_t)2 * TT * BB * NG * 2);   // reused c / m1 / m2
    bf16* xpQ  = (bf16*)take((size_t)2 * JJ * BB * NG * 2);
    bf16* Hbuf = (bf16*)take((size_t)TT * BB * H2 * 2);
    bf16* Ubuf = (bf16*)take((size_t)JJ * BB * H2 * 2);
    float* Uw2 = (float*)take((size_t)JJ * BB * 4);
    float* bw  = (float*)take((size_t)TT * BB * 4);
    bf16* htil = (bf16*)take((size_t)BB * H2 * 2);
    bf16* Util = (bf16*)take((size_t)TT * BB * H2 * 2);
    bf16* Gin  = (bf16*)take((size_t)TT * BB * G8 * 2);
    bf16* Gbuf = (bf16*)take((size_t)TT * BB * G8 * 2);
    bf16* M1b  = (bf16*)take((size_t)TT * BB * H2 * 2);
    bf16* M2b  = (bf16*)take((size_t)TT * BB * H2 * 2);
    bf16* hst  = (bf16*)take((size_t)2 * 2 * BB * DD * 2);
    float* cst = (float*)take((size_t)2 * BB * DD * 4);

    const int MR = TT * BB;  // 25600
    const int MQ = JJ * BB;  // 3072

    k_init<<<1, 64, 0, stream>>>(ctrl);

    auto cvt = [&](const float* s, bf16* d, int rows, int cols, int pad) {
        size_t total = (size_t)rows * pad;
        int blk = (int)((total + 255) / 256);
        if (blk > 8192) blk = 8192;
        k_cvtpad<<<blk, 256, 0, stream>>>(s, d, rows, cols, pad);
    };
    cvt(X, Xpad, MR, II, IPAD);
    cvt(Q, Qpad, MQ, II, IPAD);
    cvt(cW_ih, WihC, 2 * NG, II, IPAD);
    cvt(qW_ih, WihQ, 2 * NG, II, IPAD);
    cvt(cW_hh, WhhC, 2 * NG, DD, DD);
    cvt(qW_hh, WhhQ, 2 * NG, DD, DD);
    cvt(m1W_hh, Whh1, 2 * NG, DD, DD);
    cvt(m2W_hh, Whh2, 2 * NG, DD, DD);
    cvt(m1W_ih, Wih1, 2 * NG, G8, G8);
    cvt(m2W_ih, Wih2, 2 * NG, H2, H2);
    cvt(att1_w, A1W, G8, G8, G8);

    // xp for context & query bilstms (bias folded in)
    for (int dir = 0; dir < 2; ++dir) {
        k_gemm<<<(MR / 64) * (NG / 256), 256, 0, stream>>>(
            Xpad, WihC + (size_t)dir * NG * IPAD, xpB + (size_t)dir * MR * NG,
            cb_ih + dir * NG, cb_hh + dir * NG, MR, NG, IPAD);
        k_gemm<<<(MQ / 64) * (NG / 256), 256, 0, stream>>>(
            Qpad, WihQ + (size_t)dir * NG * IPAD, xpQ + (size_t)dir * MQ * NG,
            qb_ih + dir * NG, qb_hh + dir * NG, MQ, NG, IPAD);
    }
    k_lstm<<<8, 256, 0, stream>>>(xpB, WhhC, Hbuf, hst, cst, ctrl + 0, TT);
    k_lstm<<<8, 256, 0, stream>>>(xpQ, WhhQ, Ubuf, hst, cst, ctrl + 4, JJ);

    // attention
    k_uw2<<<12, 256, 0, stream>>>(Ubuf, att_w, Uw2);
    k_attn<<<TT, 256, 0, stream>>>(Hbuf, Ubuf, Uw2, att_w, att_b, bw, Util);
    k_htil<<<BB, 512, 0, stream>>>(Hbuf, bw, htil);
    k_gin<<<8192, 256, 0, stream>>>(Hbuf, Util, htil, Gin);

    // G = Gin @ att1_w^T + att1_b
    k_gemm<<<(MR / 64) * (G8 / 256), 256, 0, stream>>>(Gin, A1W, Gbuf, att1_b, nullptr,
                                                       MR, G8, G8);
    // M1 bilstm
    for (int dir = 0; dir < 2; ++dir)
        k_gemm<<<(MR / 64) * (NG / 256), 256, 0, stream>>>(
            Gbuf, Wih1 + (size_t)dir * NG * G8, xpB + (size_t)dir * MR * NG,
            m1b_ih + dir * NG, m1b_hh + dir * NG, MR, NG, G8);
    k_lstm<<<8, 256, 0, stream>>>(xpB, Whh1, M1b, hst, cst, ctrl + 8, TT);

    // M2 bilstm
    for (int dir = 0; dir < 2; ++dir)
        k_gemm<<<(MR / 64) * (NG / 256), 256, 0, stream>>>(
            M1b, Wih2 + (size_t)dir * NG * H2, xpB + (size_t)dir * MR * NG,
            m2b_ih + dir * NG, m2b_hh + dir * NG, MR, NG, H2);
    k_lstm<<<8, 256, 0, stream>>>(xpB, Whh2, M2b, hst, cst, ctrl + 12, TT);

    // outputs
    k_pout<<<TT, 256, 0, stream>>>(Gbuf, M1b, o1_w, o1_b, out);
    k_pout<<<TT, 256, 0, stream>>>(Gbuf, M2b, o2_w, o2_b, out + (size_t)TT * BB);
}
